// Generator_52338471469615
// MI455X (gfx1250) — matrix operand order repaired and checked by reading, not compiled, not run on hardware
//
#include <hip/hip_runtime.h>
#include <hip/hip_bf16.h>
#include <math.h>

// ---------------- CDNA5 WMMA types ----------------
typedef __attribute__((ext_vector_type(16))) _Float16 v16h;
typedef __attribute__((ext_vector_type(8)))  _Float16 v8h;
typedef __attribute__((ext_vector_type(8)))  float    v8f;
typedef __attribute__((ext_vector_type(4)))  int      v4i;

#define LRELU_A 0.2f
#define SQRT2_F 1.41421356237309515f

// Optional CDNA5 async global->LDS path (guarded so compile never breaks)
#if defined(__has_builtin)
#if __has_builtin(__builtin_amdgcn_global_load_async_to_lds_b128)
#define ASYNC_LDS 1
#endif
#endif

#if defined(ASYNC_LDS)
typedef __attribute__((address_space(1))) v4i gv4i;   // global (AS1) int4
typedef __attribute__((address_space(3))) v4i lv4i;   // LDS (AS3) int4
static __device__ __forceinline__ void wait_async0() {
#if __has_builtin(__builtin_amdgcn_s_wait_asynccnt)
    __builtin_amdgcn_s_wait_asynccnt(0);
#else
    asm volatile("s_wait_asynccnt 0" ::: "memory");
#endif
}
#endif

// Activation layout: NCHW32  ->  act[((b*(C/32) + c/32)*P + p)*32 + (c%32)]
// (16-byte-contiguous channel chunks; feeds b128 loads and WMMA K-tiles)

// =====================================================================
// Mapping network: PixelNorm + 8 x EqualLinear(512,512, lr_mul=0.01) with
// fused leaky-relu * sqrt2.  One workgroup (512 threads), loops layers.
// =====================================================================
__global__ __launch_bounds__(512)
void k_map(const float* __restrict__ z, const float* __restrict__ mw,
           const float* __restrict__ mb, float* __restrict__ latent)
{
    __shared__ float lat[2][512];
    __shared__ float red[512];
    int t = threadIdx.x;
    lat[0][t] = z[t];
    lat[1][t] = z[512 + t];
    __syncthreads();
    for (int b = 0; b < 2; ++b) {
        float v = lat[b][t];
        red[t] = v * v;
        __syncthreads();
        for (int s = 256; s > 0; s >>= 1) {
            if (t < s) red[t] += red[t + s];
            __syncthreads();
        }
        float inv = rsqrtf(red[0] * (1.0f / 512.0f) + 1e-8f);
        __syncthreads();
        lat[b][t] *= inv;
        __syncthreads();
    }
    const float scale = (1.0f / sqrtf(512.0f)) * 0.01f;
    for (int l = 0; l < 8; ++l) {
        const float* W = mw + ((size_t)l * 512 + t) * 512;
        float a0 = 0.f, a1 = 0.f;
        for (int j = 0; j < 512; ++j) {
            float wv = W[j];
            a0 += lat[0][j] * wv;
            a1 += lat[1][j] * wv;
        }
        float bb = mb[l * 512 + t] * 0.01f;
        float o0 = a0 * scale + bb;
        float o1 = a1 * scale + bb;
        o0 = (o0 > 0.f ? o0 : LRELU_A * o0) * SQRT2_F;
        o1 = (o1 > 0.f ? o1 : LRELU_A * o1) * SQRT2_F;
        __syncthreads();
        lat[0][t] = o0;
        lat[1][t] = o1;
        __syncthreads();
    }
    latent[t]       = lat[0][t];
    latent[512 + t] = lat[1][t];
}

// =====================================================================
// Style GEMV: s[b,i] = latent[b,:] . mod_w[i,:] / sqrt(512) + mod_b[i]
// =====================================================================
__global__ void k_style(const float* __restrict__ latent,
                        const float* __restrict__ mod_w,
                        const float* __restrict__ mod_b,
                        float* __restrict__ s, int I)
{
    int idx = blockIdx.x * blockDim.x + threadIdx.x;
    if (idx >= 2 * I) return;
    int b = idx / I, i = idx % I;
    const float ms = 0.044194173824159216f; // 1/sqrt(512)
    const float* lw = latent + b * 512;
    const float* wr = mod_w + (size_t)i * 512;
    float acc = 0.f;
    for (int j = 0; j < 512; ++j) acc += lw[j] * wr[j];
    s[idx] = acc * ms + mod_b[i];
}

// =====================================================================
// Weight modulation prep:
//   wmod[b,o,tap,i] (f16) = (1/sqrt(9I)) * w[o,i,tap] * s[b,i]
//   demod[b,o] = rsqrt(sum wm^2 + 1e-8)
// =====================================================================
__global__ __launch_bounds__(256)
void k_prep(const float* __restrict__ w, const float* __restrict__ s,
            _Float16* __restrict__ wmod, float* __restrict__ demod,
            int I, int O)
{
    int bo = blockIdx.x;          // [0, 2*O)
    int b = bo / O, o = bo % O;
    int t = threadIdx.x;
    float scale = rsqrtf((float)(I * 9));
    const float* wrow = w + (size_t)o * I * 9;
    const float* sb   = s + b * I;
    _Float16* wm      = wmod + (size_t)bo * 9 * I;
    float ss = 0.f;
    int n = I * 9;
    for (int e = t; e < n; e += 256) {
        int i = e / 9, tap = e % 9;
        float v = scale * wrow[e] * sb[i];
        ss += v * v;
        wm[(size_t)tap * I + i] = (_Float16)v;
    }
    __shared__ float red[256];
    red[t] = ss;
    __syncthreads();
    for (int sft = 128; sft > 0; sft >>= 1) {
        if (t < sft) red[t] += red[t + sft];
        __syncthreads();
    }
    if (t == 0) demod[bo] = rsqrtf(red[0] + 1e-8f);
}

// =====================================================================
// Broadcast const input [1,512,4,4] f32 -> NCHW32 f16, batch 2
// =====================================================================
__global__ void k_const(const float* __restrict__ c, _Float16* __restrict__ act)
{
    int idx = blockIdx.x * blockDim.x + threadIdx.x;
    if (idx >= 512 * 16) return;
    int ch = idx >> 4, p = idx & 15;
    _Float16 v = (_Float16)c[idx];
    size_t dst = ((size_t)(ch >> 5) * 16 + p) * 32 + (ch & 31);
    act[dst] = v;
    act[(size_t)512 * 16 + dst] = v;
}

// =====================================================================
// Implicit-GEMM modulated 3x3 conv (normal / transposed-stride-2) via WMMA.
//   per wave: D[16 out-ch x 16 pixels] f32; linearized K loop over
//   9 taps x (I/64) chunks; 2 independent v_wmma_f32_16x16x32_f16 per chunk
//   (split accumulators, no intra-chunk RAW chain).
//   B tiles double-buffered in LDS; fill for chunk k+1 issued via
//   global_load_async_to_lds_b128 and overlapped with chunk k's WMMAs;
//   per-wave s_wait_asynccnt 0 publishes LDS before the single barrier.
//   A fragments straight from global (K-contiguous wmod rows, L2 resident).
// grid: (ceil(P/16), O/64, 2), block: 128 (4 waves).
// =====================================================================
__global__ __launch_bounds__(128)
void k_conv_wmma(const _Float16* __restrict__ xin,
                 const _Float16* __restrict__ wmod,
                 const float* __restrict__ demod,
                 const float* __restrict__ noise,
                 const float* __restrict__ noise_w,
                 const float* __restrict__ act_b,
                 _Float16* __restrict__ out,
                 int I, int O, int H, int W, int OH, int OW,
                 int upsample, int fuse_act)
{
    __shared__ __attribute__((aligned(128))) _Float16 Blds[2][16][64];

    int tid  = threadIdx.x;
    int lane = tid & 31;
    int wid  = tid >> 5;
    int b    = blockIdx.z;
    int o0   = (blockIdx.y * 4 + wid) * 16;
    int P    = OH * OW;
    int Pin  = H * W;
    int pt   = blockIdx.x * 16;

    // fill assignment: 128 threads x 8 halves = 16 pixels x 64 K
    int fn  = tid >> 3;         // pixel in tile
    int fk0 = (tid & 7) * 8;    // K start within 64-chunk
    int fp  = pt + fn;
    int fy  = fp / OW, fx = fp % OW;
    bool fpv = fp < P;

    // WMMA fragment lane mapping (wave32, 16x16x32 f16)
    int mA  = lane & 15;
    int kbA = (lane < 16) ? 0 : 8;
    int nB  = lane & 15;
    int kbB = (lane < 16) ? 0 : 8;

    v8f acc0 = {0.f, 0.f, 0.f, 0.f, 0.f, 0.f, 0.f, 0.f};
    v8f acc1 = {0.f, 0.f, 0.f, 0.f, 0.f, 0.f, 0.f, 0.f};

    const _Float16* xb = xin + (size_t)b * (I >> 5) * Pin * 32;
    const _Float16* wbase = wmod + (size_t)(b * O + o0 + mA) * 9 * (size_t)I;

    int nI   = I >> 6;                   // 64-K chunks per tap (2/4/8)
    int nIsh = __builtin_ctz(nI);
    int NKK  = 9 << nIsh;                // total chunks

    // issue the B-tile fill for chunk kk into LDS buffer buf
    auto issue_fill = [&](int kk, int buf) {
        int tap = kk >> nIsh;
        int i0  = (kk & (nI - 1)) << 6;
        int ky = tap / 3, kx = tap % 3;
        int sy, sx;
        bool ok = fpv;
        if (upsample) {
            // conv_transpose(stride2) gather: xd[y+ky-2, x+kx-2], even coords
            int uy = fy + ky - 2, ux = fx + kx - 2;
            ok = ok && uy >= 0 && ux >= 0 && ((uy & 1) == 0) && ((ux & 1) == 0);
            sy = uy >> 1; sx = ux >> 1;
            ok = ok && sy < H && sx < W;
        } else {
            sy = fy + ky - 1; sx = fx + kx - 1;
            ok = ok && sy >= 0 && sy < H && sx >= 0 && sx < W;
        }
        int i = i0 + fk0;
        if (ok) {
            const _Float16* gp =
                xb + (((size_t)(i >> 5)) * Pin + (size_t)sy * W + sx) * 32 + (i & 31);
#if defined(ASYNC_LDS)
            __builtin_amdgcn_global_load_async_to_lds_b128(
                (gv4i*)gp, (lv4i*)&Blds[buf][fn][fk0], 0, 0);
#else
            *(v8h*)&Blds[buf][fn][fk0] = *(const v8h*)gp;
#endif
        } else {
            v8h zz = {(_Float16)0, (_Float16)0, (_Float16)0, (_Float16)0,
                      (_Float16)0, (_Float16)0, (_Float16)0, (_Float16)0};
            *(v8h*)&Blds[buf][fn][fk0] = zz;
        }
    };

    // software pipeline: one barrier per chunk, fill(k+1) overlaps compute(k)
    issue_fill(0, 0);
#if defined(ASYNC_LDS)
    wait_async0();
#endif
    __syncthreads();

    for (int kk = 0; kk < NKK; ++kk) {
        int cur = kk & 1;
        if (kk + 1 < NKK) issue_fill(kk + 1, cur ^ 1);

        int tap = kk >> nIsh;
        int i0  = (kk & (nI - 1)) << 6;
        int wtap = upsample ? (8 - tap) : tap;   // spatial flip for transpose
        const _Float16* wrow = wbase + (size_t)wtap * I;

        union { v16h v; v8h h[2]; } A0, A1, B0, B1;
        A0.h[0] = *(const v8h*)(wrow + i0 + kbA);
        A0.h[1] = *(const v8h*)(wrow + i0 + kbA + 16);
        A1.h[0] = *(const v8h*)(wrow + i0 + 32 + kbA);
        A1.h[1] = *(const v8h*)(wrow + i0 + 32 + kbA + 16);
        B0.h[0] = *(const v8h*)(&Blds[cur][nB][kbB]);
        B0.h[1] = *(const v8h*)(&Blds[cur][nB][kbB + 16]);
        B1.h[0] = *(const v8h*)(&Blds[cur][nB][32 + kbB]);
        B1.h[1] = *(const v8h*)(&Blds[cur][nB][32 + kbB + 16]);

        acc0 = __builtin_amdgcn_wmma_f32_16x16x32_f16(
            false, A0.v, false, B0.v, (short)0, acc0, false, false);
        acc1 = __builtin_amdgcn_wmma_f32_16x16x32_f16(
            false, A1.v, false, B1.v, (short)0, acc1, false, false);

#if defined(ASYNC_LDS)
        wait_async0();     // publish my async fill before the barrier
#endif
        __syncthreads();
    }

    // epilogue: *demod (+noise +bias, lrelu*sqrt2) -> one b128 f16 store
    int p = pt + nB;
    bool pv = p < P;
    int mbase = (lane < 16) ? 0 : 8;
    float nv = 0.f, nw = 0.f;
    if (fuse_act && pv) {
        nv = noise[(size_t)b * P + p];
        nw = noise_w[0];
    }
    v8h ov;
#pragma unroll
    for (int j = 0; j < 8; ++j) {
        int o = o0 + mbase + j;
        float v = (acc0[j] + acc1[j]) * demod[b * O + o];
        if (fuse_act) {
            v += nw * nv + act_b[o];
            v = (v > 0.f ? v : LRELU_A * v) * SQRT2_F;
        }
        ov[j] = (_Float16)v;
    }
    if (pv) {
        int o = o0 + mbase;
        size_t dst = (((size_t)b * (O >> 5) + (o >> 5)) * (size_t)P + p) * 32 + (o & 31);
        *(v8h*)(out + dst) = ov;
    }
}

// =====================================================================
// Post-upsample 4x4 FIR blur (pad 1,1) on (r+1)^2 -> r^2, fused with
// noise add + bias + lrelu*sqrt2.  NCHW32 f16 in/out.
// =====================================================================
__global__ void k_blur(const _Float16* __restrict__ tmp,
                       const float* __restrict__ noise,
                       const float* __restrict__ noise_w,
                       const float* __restrict__ act_b,
                       _Float16* __restrict__ out, int C, int r)
{
    int idx = blockIdx.x * blockDim.x + threadIdx.x;
    int total = 2 * C * r * r;
    if (idx >= total) return;
    int ci = idx & 31;
    int p  = (idx >> 5) % (r * r);
    int cc = ((idx >> 5) / (r * r)) % (C >> 5);
    int b  = idx / (C * r * r);
    int y = p / r, x = p % r;
    int S = r + 1;
    const float K4[4] = {1.f, 3.f, 3.f, 1.f};
    const _Float16* tb =
        tmp + (((size_t)b * (C >> 5) + cc) * (size_t)S * S) * 32 + ci;
    float acc = 0.f;
#pragma unroll
    for (int u = 0; u < 4; ++u) {
        int sy = y + u - 1;
        if (sy < 0 || sy >= S) continue;
#pragma unroll
        for (int v = 0; v < 4; ++v) {
            int sx = x + v - 1;
            if (sx < 0 || sx >= S) continue;
            acc += K4[u] * K4[v] * (float)tb[(size_t)(sy * S + sx) * 32];
        }
    }
    acc *= (1.f / 16.f);   // outer([1,3,3,1])/64 * 4
    acc += noise_w[0] * noise[(size_t)b * r * r + p] + act_b[cc * 32 + ci];
    acc = (acc > 0.f ? acc : LRELU_A * acc) * SQRT2_F;
    out[idx] = (_Float16)acc;
}

// =====================================================================
// toRGB: 1x1 modulated conv (no demod) + bias + inline skip-upsample-blur.
// x in NCHW32 f16; skip_prev fp32 at res r/2; out fp32 [2,3,r,r].
// =====================================================================
__global__ void k_rgb(const _Float16* __restrict__ x,
                      const float* __restrict__ w,
                      const float* __restrict__ s,
                      const float* __restrict__ bias,
                      const float* __restrict__ skip,
                      float* __restrict__ out,
                      int I, int r, int has_skip)
{
    int idx = blockIdx.x * blockDim.x + threadIdx.x;
    int total = 2 * 3 * r * r;
    if (idx >= total) return;
    int px = idx % (r * r);
    int c  = (idx / (r * r)) % 3;
    int b  = idx / (3 * r * r);
    float scale = rsqrtf((float)I);
    const _Float16* xb = x + (size_t)b * I * r * r;  // batch stride = I*P
    const float* wc = w + c * I;
    const float* sb = s + b * I;
    float acc = 0.f;
    for (int cc = 0; cc < (I >> 5); ++cc) {
        const _Float16* xp = xb + ((size_t)cc * r * r + px) * 32;
        int ib = cc * 32;
#pragma unroll
        for (int ci = 0; ci < 32; ++ci)
            acc += wc[ib + ci] * sb[ib + ci] * (float)xp[ci];
    }
    acc = acc * scale + bias[c];
    if (has_skip) {
        int y = px / r, xx = px % r;
        int rp = r >> 1;
        const float* sp = skip + (size_t)(b * 3 + c) * rp * rp;
        const float K4[4] = {1.f, 3.f, 3.f, 1.f};
        float sacc = 0.f;
#pragma unroll
        for (int u = 0; u < 4; ++u) {
            int uy = y + u - 2;
            if (uy < 0 || (uy & 1) || uy > 2 * (rp - 1)) continue;
#pragma unroll
            for (int v = 0; v < 4; ++v) {
                int ux = xx + v - 2;
                if (ux < 0 || (ux & 1) || ux > 2 * (rp - 1)) continue;
                sacc += K4[u] * K4[v] * sp[(uy >> 1) * rp + (ux >> 1)];
            }
        }
        acc += sacc * (1.f / 16.f);
    }
    out[idx] = acc;
}

// =====================================================================
// Host orchestration
// =====================================================================
extern "C" void kernel_launch(void* const* d_in, const int* in_sizes, int n_in,
                              void* d_out, int out_size, void* d_ws, size_t ws_size,
                              hipStream_t stream)
{
    (void)in_sizes; (void)n_in; (void)out_size; (void)ws_size;

    // input layout (setup_inputs dict order):
    // 0:z  1:mapping.w  2:mapping.b  3:const
    // 4 + 5*ci + {0:w,1:mod_w,2:mod_b,3:act_b,4:noise_w}   ci in [0,13)
    // 69 + 4*ti + {0:w,1:mod_w,2:mod_b,3:bias}             ti in [0,7)
    // 97 + ni : noises                                     ni in [0,13)
    const float* z     = (const float*)d_in[0];
    const float* map_w = (const float*)d_in[1];
    const float* map_b = (const float*)d_in[2];
    const float* cst   = (const float*)d_in[3];
    auto conv_in  = [&](int ci, int f) { return (const float*)d_in[4 + 5 * ci + f]; };
    auto rgb_in   = [&](int ti, int f) { return (const float*)d_in[69 + 4 * ti + f]; };
    auto noise_in = [&](int ni)        { return (const float*)d_in[97 + ni]; };

    static const int ICH[13] = {512,512,512,512,512,512,512,512,512,512,256,256,128};
    static const int OCH[13] = {512,512,512,512,512,512,512,512,512,256,256,128,128};

    // workspace carve-up (~119 MiB)
    char* ws = (char*)d_ws;
    float*    latent = (float*)(ws);                       // 4 KB
    float*    sbuf   = (float*)(ws + (4  << 10));          // 4 KB
    float*    demod  = (float*)(ws + (8  << 10));          // 4 KB
    _Float16* wmod   = (_Float16*)(ws + (16 << 10));       // <= 9.44 MB
    const size_t ACT = 36ull << 20;
    _Float16* actA = (_Float16*)(ws + (10ull << 20));
    _Float16* actB = (_Float16*)(ws + (10ull << 20) + ACT);
    _Float16* tmpb = (_Float16*)(ws + (10ull << 20) + 2 * ACT);
    float*    skipA = (float*)(ws + (10ull << 20) + 3 * ACT);
    float*    skipB = skipA + 2 * 3 * 128 * 128;

    auto launch_conv = [&](int ci, const _Float16* in, _Float16* outp,
                           int rin, int upsample, int fuse, const float* noise) {
        int I = ICH[ci], O = OCH[ci];
        int n = 2 * I;
        k_style<<<(n + 255) / 256, 256, 0, stream>>>(latent, conv_in(ci, 1), conv_in(ci, 2), sbuf, I);
        k_prep<<<2 * O, 256, 0, stream>>>(conv_in(ci, 0), sbuf, wmod, demod, I, O);
        int OHW = upsample ? (2 * rin + 1) : rin;
        int P = OHW * OHW;
        dim3 g((P + 15) / 16, O / 64, 2);
        k_conv_wmma<<<g, 128, 0, stream>>>(in, wmod, demod, noise, conv_in(ci, 4), conv_in(ci, 3),
                                           outp, I, O, rin, rin, OHW, OHW, upsample, fuse);
    };
    auto launch_rgb = [&](int ti, const _Float16* x, const float* skip, float* outp,
                          int I, int r, int has_skip) {
        k_style<<<(2 * I + 255) / 256, 256, 0, stream>>>(latent, rgb_in(ti, 1), rgb_in(ti, 2), sbuf, I);
        int tot = 2 * 3 * r * r;
        k_rgb<<<(tot + 255) / 256, 256, 0, stream>>>(x, rgb_in(ti, 0), sbuf, rgb_in(ti, 3),
                                                     skip, outp, I, r, has_skip);
    };

    // 1) mapping network
    k_map<<<1, 512, 0, stream>>>(z, map_w, map_b, latent);
    // 2) const input -> actA (f16 NCHW32, broadcast to batch 2)
    k_const<<<(512 * 16 + 255) / 256, 256, 0, stream>>>(cst, actA);
    // 3) conv0 @4x4: actA -> actB, fused act+noise0
    launch_conv(0, actA, actB, 4, 0, 1, noise_in(0));
    // 4) toRGB0 -> skipA
    launch_rgb(0, actB, nullptr, skipA, 512, 4, 0);

    float* skips[2] = {skipA, skipB};
    int cur = 0;
    for (int t = 1; t <= 6; ++t) {
        int r = 4 << t, rin = r >> 1;
        int ciU = 2 * t - 1, ciN = 2 * t;
        // upsample conv (transpose stride2): actB -> tmpb at (r+1)^2, no act fuse
        launch_conv(ciU, actB, tmpb, rin, 1, 0, nullptr);
        // blur + noise + bias + act: tmpb -> actA at r^2
        int C = OCH[ciU];
        int tot = 2 * C * r * r;
        k_blur<<<(tot + 255) / 256, 256, 0, stream>>>(tmpb, noise_in(2 * t - 1),
                                                      conv_in(ciU, 4), conv_in(ciU, 3),
                                                      actA, C, r);
        // normal conv: actA -> actB, fused
        launch_conv(ciN, actA, actB, r, 0, 1, noise_in(2 * t));
        // toRGB with skip upsample; final stage writes d_out (fp32)
        float* outp = (t == 6) ? (float*)d_out : skips[cur ^ 1];
        launch_rgb(t, actB, skips[cur], outp, OCH[ciN], r, 1);
        cur ^= 1;
    }
}
